// Cross_Atten_Lite_split_63995012710843
// MI455X (gfx1250) — matrix operand order repaired and checked by reading, not compiled, not run on hardware
//
#include <hip/hip_runtime.h>

typedef __attribute__((ext_vector_type(8)))  _Float16 v8h;
typedef __attribute__((ext_vector_type(16))) _Float16 v16h;
typedef __attribute__((ext_vector_type(8)))  float    v8f;

// Problem dims (fixed by the reference)
// B=4, C=256, H=W=64 -> N=4096 tokens/batch, D=64 head dim, 2 heads

// ---- workspace layout (bytes) ----
static constexpr size_t WK1_OFF  = 0;        // kq1_w f16: 128x256
static constexpr size_t WK2_OFF  = 65536;    // kq2_w f16: 128x256
static constexpr size_t WV_OFF   = 131072;   // v_w   f16: 128x256
static constexpr size_t WOUT_OFF = 196608;   // out_w f16: 256x128
static constexpr size_t BNP_OFF  = 262144;   // 1536 f32: s1,sh1,s2,sh2,sl,shl
static constexpr size_t Q_OFF    = 524288;   // f16 [b][h][n][64]   : 4 MiB
static constexpr size_t K_OFF    = Q_OFF + 4194304;  // f16 [b][h][n][64]
static constexpr size_t V_OFF    = K_OFF + 4194304;  // f16 [b][h][64][n]  (transposed)
static constexpr size_t O_OFF    = V_OFF + 4194304;  // f16 [b][n][128]

// ---- WMMA fragment loaders (gfx1250 16-bit layouts, ISA 7.12.2) ----
// A (16x32, MxK) from row-major [row][K], ldk halves; 16B-aligned rows required.
__device__ __forceinline__ v16h ldfrag_a(const _Float16* base, int ldk) {
  const int lane = threadIdx.x & 31;
  const _Float16* p = base + (lane & 15) * ldk + ((lane >> 4) << 3);
  v8h lo = *(const v8h*)(p);        // K = 8*hi + 0..7
  v8h hi16 = *(const v8h*)(p + 16); // K = 16 + 8*hi + 0..7
  return __builtin_shufflevector(lo, hi16, 0,1,2,3,4,5,6,7,8,9,10,11,12,13,14,15);
}

// B (32x16, KxN) from row-major [N][K] (K-contiguous), ldk halves.
__device__ __forceinline__ v16h ldfrag_b(const _Float16* base, int ldk) {
  const int lane = threadIdx.x & 31;
  const _Float16* p = base + (lane & 15) * ldk + ((lane >> 4) << 3);
  v8h lo = *(const v8h*)(p);        // K = 8*hi + 0..7
  v8h hi16 = *(const v8h*)(p + 16); // K = 16 + 8*hi + 0..7
  return __builtin_shufflevector(lo, hi16, 0,1,2,3,4,5,6,7,8,9,10,11,12,13,14,15);
}

__device__ __forceinline__ v8f wmma_f16(v16h a, v16h b, v8f c) {
  return __builtin_amdgcn_wmma_f32_16x16x32_f16(false, a, false, b, (short)0, c, false, false);
}

// ---------------------------------------------------------------------------
// Kernel 0: weights fp32->fp16, fold BatchNorms into scale/shift
// ---------------------------------------------------------------------------
__global__ __launch_bounds__(256) void prep_kernel(
    const float* __restrict__ kq1_w, const float* __restrict__ kq2_w,
    const float* __restrict__ v_w,   const float* __restrict__ out_w,
    const float* __restrict__ g1, const float* __restrict__ b1,
    const float* __restrict__ m1, const float* __restrict__ v1,
    const float* __restrict__ g2, const float* __restrict__ b2,
    const float* __restrict__ m2, const float* __restrict__ v2,
    const float* __restrict__ gl, const float* __restrict__ bl,
    const float* __restrict__ ml, const float* __restrict__ vl,
    char* __restrict__ ws) {
  _Float16* wk1  = (_Float16*)(ws + WK1_OFF);
  _Float16* wk2  = (_Float16*)(ws + WK2_OFF);
  _Float16* wv   = (_Float16*)(ws + WV_OFF);
  _Float16* wout = (_Float16*)(ws + WOUT_OFF);
  float*    bnp  = (float*)(ws + BNP_OFF);

  const int tid = blockIdx.x * 256 + threadIdx.x;
  if (tid < 32768)       wk1[tid]          = (_Float16)kq1_w[tid];
  else if (tid < 65536)  wk2[tid - 32768]  = (_Float16)kq2_w[tid - 32768];
  else if (tid < 98304)  wv[tid - 65536]   = (_Float16)v_w[tid - 65536];
  else if (tid < 131072) wout[tid - 98304] = (_Float16)out_w[tid - 98304];

  if (tid < 256) {
    const int c = tid;
    float s = g1[c] * rsqrtf(v1[c] + 1e-5f);
    bnp[c]        = s;  bnp[256 + c]  = b1[c] - m1[c] * s;
    s = g2[c] * rsqrtf(v2[c] + 1e-5f);
    bnp[512 + c]  = s;  bnp[768 + c]  = b2[c] - m2[c] * s;
    s = gl[c] * rsqrtf(vl[c] + 1e-5f);
    bnp[1024 + c] = s;  bnp[1280 + c] = bl[c] - ml[c] * s;
  }
}

// ---------------------------------------------------------------------------
// Kernel 1: BN + projections (kq1, kq2, v), channel-shuffle folded into the
// scatter stores. Q gets the 1/sqrt(64) softmax scale folded in.
// The ms loop is fully unrolled so the weight-pointer selects constant-fold
// and the B-fragment loads lower to global_load_b128 (not flat_load).
// ---------------------------------------------------------------------------
__global__ __launch_bounds__(256) void proj_kernel(
    const float* __restrict__ x, const float* __restrict__ x1,
    const float* __restrict__ x2,
    const _Float16* __restrict__ wk1, const _Float16* __restrict__ wk2,
    const _Float16* __restrict__ wv,
    const float* __restrict__ kq1_b, const float* __restrict__ kq2_b,
    const float* __restrict__ v_b, const float* __restrict__ bnp,
    _Float16* __restrict__ Q, _Float16* __restrict__ K,
    _Float16* __restrict__ V) {
  __shared__ __attribute__((aligned(16))) _Float16 tA[64][264];  // 64 tok x 256c, padded

  const int t = threadIdx.x;
  const int t0 = blockIdx.x * 64;      // flattened (b*4096+n) token base
  const int b  = t0 >> 12;
  const int nb = t0 & 4095;
  const int lane = t & 31, w = t >> 5;
  const int rf = w & 3, ch = w >> 2;   // wave -> (row-frag, column half)
  const int hi = lane >> 4;

  const float*    srcs[3] = {x1, x2, x};
  const _Float16* wgt[3]  = {wk1, wk2, wv};
  const float*    bias[3] = {kq1_b, kq2_b, v_b};

  #pragma unroll
  for (int ms = 0; ms < 3; ++ms) {
    __syncthreads();
    {  // stage 64x256 activation tile (BN applied for x1/x2)
      const float* src = srcs[ms];
      const float* sc = bnp + ms * 512;
      const int tok = t & 63, c0 = t >> 6;
      for (int c = c0; c < 256; c += 4) {
        float vv = src[((size_t)(b * 256 + c)) * 4096 + nb + tok];
        if (ms < 2) vv = vv * sc[c] + sc[256 + c];
        tA[tok][c] = (_Float16)vv;
      }
    }
    __syncthreads();

    const _Float16* W = wgt[ms];
    #pragma unroll
    for (int cfi = 0; cfi < 4; ++cfi) {
      const int cf = ch * 4 + cfi;          // 8 col-frags of 16 over 128 outputs
      const int fb = cf * 16;
      const int f  = fb + (lane & 15);
      v8f acc = {};
      #pragma unroll
      for (int ks = 0; ks < 8; ++ks) {      // K = 256 in steps of 32
        v16h a  = ldfrag_a(&tA[16 * rf][32 * ks], 264);
        v16h bf = ldfrag_b(W + fb * 256 + 32 * ks, 256);
        acc = wmma_f16(a, bf, acc);
      }
      const float bv = bias[ms][f];

      if (ms < 2) {
        // channel shuffle: kq1[f] -> c=2f, kq2[f] -> c=2f+1; split into k1,q1,k2,q2
        const int g = f >> 5, j = f & 31;
        const int d = 2 * j + ms;
        _Float16* dst = (g & 1) ? Q : K;
        const int hsel = g >> 1;
        const float qs = (g & 1) ? 0.125f : 1.0f;  // softmax scale folded into Q
        _Float16* dp = dst + ((size_t)(b * 2 + hsel)) * 4096 * 64 + d;
        #pragma unroll
        for (int r = 0; r < 8; ++r) {
          const int n = nb + 16 * rf + r + hi * 8;
          dp[(size_t)n * 64] = (_Float16)((acc[r] + bv) * qs);
        }
      } else {
        // v1/v2 -> V stored transposed [b][h][d][n] for the PV B-operand
        const int hsel = f >> 6, d = f & 63;
        _Float16* dp = V + (((size_t)(b * 2 + hsel)) * 64 + d) * 4096;
        #pragma unroll
        for (int r = 0; r < 8; ++r) {
          const int n = nb + 16 * rf + r + hi * 8;
          dp[n] = (_Float16)(acc[r] + bv);
        }
      }
    }
  }
}

// ---------------------------------------------------------------------------
// Kernel 2: flash attention, N=4096, D=64, 8 (batch,head) problems.
// Block = 128 queries (8 waves x 16). Double-buffered 64-key chunks:
// global loads for chunk kc+1 are issued before the WMMA work on chunk kc,
// LDS stores land after compute, one barrier per iteration. Chunk kc+2 is
// prefetched (global_prefetch_b8).
// ---------------------------------------------------------------------------
__global__ __launch_bounds__(256) void attn_kernel(
    const _Float16* __restrict__ Q, const _Float16* __restrict__ K,
    const _Float16* __restrict__ V, _Float16* __restrict__ O) {
  __shared__ __attribute__((aligned(16))) _Float16 Kt[2][64][80];   // [key][d]
  __shared__ __attribute__((aligned(16))) _Float16 Vt[2][64][80];   // [d][key]
  __shared__ __attribute__((aligned(16))) _Float16 Ps[8][16][80];   // per-wave P

  const int t = threadIdx.x, lane = t & 31, w = t >> 5, hi = lane >> 4;
  const int bh = blockIdx.x >> 5, qb = blockIdx.x & 31;
  const int b = bh >> 1, h = bh & 1;
  const int qrow = qb * 128 + w * 16;

  // Q fragments live in registers for the whole kernel (scale pre-folded)
  const _Float16* Qb = Q + ((size_t)bh * 4096 + qrow) * 64;
  const v16h qa0 = ldfrag_a(Qb, 64);
  const v16h qa1 = ldfrag_a(Qb + 32, 64);

  v8f oacc[4] = {};
  float mr[8], lr[8];
  #pragma unroll
  for (int r = 0; r < 8; ++r) { mr[r] = -1e30f; lr[r] = 0.0f; }

  const _Float16* Kbase = K + (size_t)bh * 4096 * 64;
  const _Float16* Vbase = V + (size_t)bh * 64 * 4096;

  // per-thread staging coordinates: thread covers rows row0 and row0+32
  const int row0 = t >> 3, c80 = (t & 7) << 3;

  v8h kra, krb, vra, vrb;  // in-flight staging registers
  // preload chunk 0
  kra = *(const v8h*)(Kbase + ((size_t)(row0)) * 64 + c80);
  krb = *(const v8h*)(Kbase + ((size_t)(row0 + 32)) * 64 + c80);
  vra = *(const v8h*)(Vbase + (size_t)row0 * 4096 + c80);
  vrb = *(const v8h*)(Vbase + (size_t)(row0 + 32) * 4096 + c80);
  *(v8h*)&Kt[0][row0][c80]      = kra;
  *(v8h*)&Kt[0][row0 + 32][c80] = krb;
  *(v8h*)&Vt[0][row0][c80]      = vra;
  *(v8h*)&Vt[0][row0 + 32][c80] = vrb;
  __syncthreads();

  for (int kc = 0; kc < 64; ++kc) {
    const int cur = kc & 1;
    const bool more = (kc + 1) < 64;

    if (more) {  // issue global loads for the next chunk early
      const int kn = (kc + 1) * 64;
      kra = *(const v8h*)(Kbase + ((size_t)(kn + row0)) * 64 + c80);
      krb = *(const v8h*)(Kbase + ((size_t)(kn + row0 + 32)) * 64 + c80);
      vra = *(const v8h*)(Vbase + (size_t)row0 * 4096 + kn + c80);
      vrb = *(const v8h*)(Vbase + (size_t)(row0 + 32) * 4096 + kn + c80);
    }
    if (kc + 2 < 64) {  // prefetch chunk kc+2 into cache (global_prefetch_b8)
      const int kp = (kc + 2) * 64;
      __builtin_prefetch(Kbase + ((size_t)(kp + row0)) * 64 + c80, 0, 1);
      __builtin_prefetch(Vbase + (size_t)row0 * 4096 + kp + c80, 0, 1);
    }

    // S = Q K^T for 4 key-frags of 16 (K-dim = 64 in two WMMA steps)
    v8f s[4];
    #pragma unroll
    for (int kf = 0; kf < 4; ++kf) {
      v16h b0 = ldfrag_b(&Kt[cur][16 * kf][0], 80);
      v16h b1 = ldfrag_b(&Kt[cur][16 * kf][32], 80);
      v8f sv = {};
      sv = wmma_f16(qa0, b0, sv);
      sv = wmma_f16(qa1, b1, sv);
      s[kf] = sv;
    }

    // online softmax; row r's 16 columns live in the 16 lanes of this half-wave
    #pragma unroll
    for (int r = 0; r < 8; ++r) {
      float rm = fmaxf(fmaxf((float)s[0][r], (float)s[1][r]),
                       fmaxf((float)s[2][r], (float)s[3][r]));
      rm = fmaxf(rm, __shfl_xor(rm, 1));
      rm = fmaxf(rm, __shfl_xor(rm, 2));
      rm = fmaxf(rm, __shfl_xor(rm, 4));
      rm = fmaxf(rm, __shfl_xor(rm, 8));
      const float nm = fmaxf(mr[r], rm);
      const float alpha = __expf(mr[r] - nm);
      mr[r] = nm;
      float ps = 0.0f;
      #pragma unroll
      for (int kf = 0; kf < 4; ++kf) {
        const float p = __expf((float)s[kf][r] - nm);
        ps += p;
        Ps[w][r + hi * 8][16 * kf + (lane & 15)] = (_Float16)p;
      }
      ps += __shfl_xor(ps, 1);
      ps += __shfl_xor(ps, 2);
      ps += __shfl_xor(ps, 4);
      ps += __shfl_xor(ps, 8);
      lr[r] = lr[r] * alpha + ps;
      #pragma unroll
      for (int cf = 0; cf < 4; ++cf) oacc[cf][r] *= alpha;
    }

    // own-wave LDS RAW: make P visible before A-fragment reload
    asm volatile("s_wait_dscnt 0x0" ::: "memory");

    // O += P V  (A = P 16x32 from LDS, B = V^T [d][key] from LDS)
    #pragma unroll
    for (int sub = 0; sub < 2; ++sub) {
      v16h pa = ldfrag_a(&Ps[w][0][32 * sub], 80);
      #pragma unroll
      for (int cf = 0; cf < 4; ++cf) {
        v16h vb = ldfrag_b(&Vt[cur][16 * cf][32 * sub], 80);
        oacc[cf] = wmma_f16(pa, vb, oacc[cf]);
      }
    }

    if (more) {  // land the next chunk in the alternate buffer
      const int nxt = cur ^ 1;
      *(v8h*)&Kt[nxt][row0][c80]      = kra;
      *(v8h*)&Kt[nxt][row0 + 32][c80] = krb;
      *(v8h*)&Vt[nxt][row0][c80]      = vra;
      *(v8h*)&Vt[nxt][row0 + 32][c80] = vrb;
    }
    __syncthreads();
  }

  // epilogue: O/l, write f16 [b][n][128] with head-concat feature index
  #pragma unroll
  for (int cf = 0; cf < 4; ++cf) {
    const int f = h * 64 + 16 * cf + (lane & 15);
    #pragma unroll
    for (int r = 0; r < 8; ++r) {
      const int n = qrow + r + hi * 8;
      O[((size_t)(b * 4096 + n)) * 128 + f] = (_Float16)(oacc[cf][r] / lr[r]);
    }
  }
}

// ---------------------------------------------------------------------------
// Kernel 3: output projection + bias + BN affine + w_scale*o + x (NCHW f32)
// ---------------------------------------------------------------------------
__global__ __launch_bounds__(256) void outproj_kernel(
    const _Float16* __restrict__ O, const _Float16* __restrict__ wout,
    const float* __restrict__ out_b, const float* __restrict__ bnp,
    const float* __restrict__ w_scale, const float* __restrict__ x,
    float* __restrict__ out) {
  __shared__ __attribute__((aligned(16))) _Float16 Ot[64][136];
  const int t = threadIdx.x, lane = t & 31, w = t >> 5, hi = lane >> 4;
  const int t0 = blockIdx.x * 64, b = t0 >> 12, nb = t0 & 4095;

  #pragma unroll
  for (int it = 0; it < 4; ++it) {  // stage 64x128 O tile
    const int i = t + it * 256;
    const int row = i >> 4, c8 = (i & 15) << 3;
    *(v8h*)&Ot[row][c8] = *(const v8h*)(O + ((size_t)(t0 + row)) * 128 + c8);
  }
  __syncthreads();

  const int rf = w & 3, ch = w >> 2;
  const float wsc = w_scale[0];
  const float* sl = bnp + 1024;
  const float* shl = bnp + 1280;

  #pragma unroll
  for (int cfi = 0; cfi < 8; ++cfi) {
    const int cf = ch * 8 + cfi;          // 16 col-frags over 256 outputs
    const int c = 16 * cf + (lane & 15);
    v8f acc = {};
    #pragma unroll
    for (int ks = 0; ks < 4; ++ks) {      // K = 128
      v16h a  = ldfrag_a(&Ot[16 * rf][32 * ks], 136);
      v16h bf = ldfrag_b(wout + (16 * cf) * 128 + 32 * ks, 128);
      acc = wmma_f16(a, bf, acc);
    }
    const float obv = out_b[c], bs = sl[c], bsh = shl[c];
    #pragma unroll
    for (int r = 0; r < 8; ++r) {
      const int n = nb + 16 * rf + r + hi * 8;
      const size_t gi = ((size_t)(b * 256 + c)) * 4096 + n;
      out[gi] = wsc * ((acc[r] + obv) * bs + bsh) + x[gi];
    }
  }
}

// ---------------------------------------------------------------------------
extern "C" void kernel_launch(void* const* d_in, const int* in_sizes, int n_in,
                              void* d_out, int out_size, void* d_ws, size_t ws_size,
                              hipStream_t stream) {
  (void)in_sizes; (void)n_in; (void)out_size; (void)ws_size;
  const float* x     = (const float*)d_in[0];
  const float* x1    = (const float*)d_in[1];
  const float* x2    = (const float*)d_in[2];
  const float* bn1_g = (const float*)d_in[3];
  const float* bn1_b = (const float*)d_in[4];
  const float* bn1_m = (const float*)d_in[5];
  const float* bn1_v = (const float*)d_in[6];
  const float* bn2_g = (const float*)d_in[7];
  const float* bn2_b = (const float*)d_in[8];
  const float* bn2_m = (const float*)d_in[9];
  const float* bn2_v = (const float*)d_in[10];
  const float* kq1_w = (const float*)d_in[11];
  const float* kq1_b = (const float*)d_in[12];
  const float* kq2_w = (const float*)d_in[13];
  const float* kq2_b = (const float*)d_in[14];
  const float* v_w   = (const float*)d_in[15];
  const float* v_b   = (const float*)d_in[16];
  const float* out_w = (const float*)d_in[17];
  const float* out_b = (const float*)d_in[18];
  const float* bnl_g = (const float*)d_in[19];
  const float* bnl_b = (const float*)d_in[20];
  const float* bnl_m = (const float*)d_in[21];
  const float* bnl_v = (const float*)d_in[22];
  const float* w_scale = (const float*)d_in[23];

  char* ws = (char*)d_ws;
  _Float16* Qp = (_Float16*)(ws + Q_OFF);
  _Float16* Kp = (_Float16*)(ws + K_OFF);
  _Float16* Vp = (_Float16*)(ws + V_OFF);
  _Float16* Op = (_Float16*)(ws + O_OFF);
  const _Float16* wk1p  = (const _Float16*)(ws + WK1_OFF);
  const _Float16* wk2p  = (const _Float16*)(ws + WK2_OFF);
  const _Float16* wvp   = (const _Float16*)(ws + WV_OFF);
  const _Float16* woutp = (const _Float16*)(ws + WOUT_OFF);
  const float*    bnpp  = (const float*)(ws + BNP_OFF);

  prep_kernel<<<512, 256, 0, stream>>>(kq1_w, kq2_w, v_w, out_w,
                                       bn1_g, bn1_b, bn1_m, bn1_v,
                                       bn2_g, bn2_b, bn2_m, bn2_v,
                                       bnl_g, bnl_b, bnl_m, bnl_v, ws);
  proj_kernel<<<256, 256, 0, stream>>>(x, x1, x2, wk1p, wk2p, wvp,
                                       kq1_b, kq2_b, v_b, bnpp, Qp, Kp, Vp);
  attn_kernel<<<256, 256, 0, stream>>>(Qp, Kp, Vp, Op);
  outproj_kernel<<<256, 256, 0, stream>>>(Op, woutp, out_b, bnpp, w_scale, x,
                                          (float*)d_out);
}
